// PoseMLP_24429773979877
// MI455X (gfx1250) — compile-verified, matrix operand order checked by reading, not run on hardware
//
#include <hip/hip_runtime.h>
#include <hip/hip_bf16.h>

// ---------------------------------------------------------------------------
// PoseMLP fused 3-layer grouped MLP for MI455X (gfx1250, wave32, WMMA)
//   h1 = relu(X @ W1 + b1)   [B,24,256]x[24,256,128]
//   h2 = relu(h1 @ W2 + b2)  [24,128,128]
//   y  = h2 @ W3 + b3        [24,128,6]
// Memory-bound (402 MB pose_feat stream vs 39 GFLOP): weights packed once to
// f16 in WMMA-B-fragment order (2.4 MB, L2-resident), activations streamed
// f32->f16 into LDS, whole MLP fused per 32-row/1-joint tile with
// v_wmma_f32_16x16x32_f16. Weight staging uses gfx1250 async global->LDS DMA
// (ASYNCcnt) overlapped with input conversion / layer epilogues.
// ---------------------------------------------------------------------------

typedef __attribute__((ext_vector_type(16))) _Float16 v16h;
typedef __attribute__((ext_vector_type(8)))  _Float16 v8h;
typedef __attribute__((ext_vector_type(8)))  float    v8f;

#define NJ   24
#define IND  256
#define HID  128
#define OUTD 6
#define OUTP 16          // padded out dim (one 16-wide WMMA tile)
#define BATCH 16384

#define MT   32          // batch rows per block
#define SA   (IND + 8)   // padded LDS stride for input acts (bank-conflict-free)
#define SH   (HID + 8)   // padded LDS stride for hidden acts

// workspace layout (in _Float16 elements)
#define W1S_ELEMS ((size_t)NJ * IND * HID)                 // 786432
#define W2S_ELEMS ((size_t)NJ * HID * HID)                 // 393216
#define W3S_ELEMS ((size_t)NJ * HID * OUTP)                // 49152
#define OFF_W2 (W1S_ELEMS)
#define OFF_W3 (W1S_ELEMS + W2S_ELEMS)

// ---------------------------------------------------------------------------
// Weight conversion: f32 [J][K][N] -> f16 swizzled [J][kc][nt][lane][e]
// Fragment layout mirrors the 16-bit 32xK WMMA operand layout:
//   lo = lane&15, hi = lane>>4 ; n = nt*16+lo
//   e<8 : k = kc*32 + hi*8 + e ; e>=8 : k = kc*32 + 16 + hi*8 + (e-8)
// ---------------------------------------------------------------------------
__global__ void cvt_swz(const float* __restrict__ src, _Float16* __restrict__ dst,
                        int K, int N, int KN, int total) {
  int tid = blockIdx.x * blockDim.x + threadIdx.x;
  if (tid >= total) return;
  int j      = tid / KN;
  int d      = tid - j * KN;
  int ntN    = N >> 4;
  int tile   = d >> 9;          // 512 elems per 32x16 tile
  int within = d & 511;
  int lane   = within >> 4;
  int e      = within & 15;
  int kc     = tile / ntN;
  int nt     = tile - kc * ntN;
  int lo = lane & 15, hi = lane >> 4;
  int n  = nt * 16 + lo;
  int kin = (e < 8) ? (hi * 8 + e) : (16 + hi * 8 + (e - 8));
  int k  = kc * 32 + kin;
  dst[tid] = (_Float16)src[(size_t)j * KN + (size_t)k * N + n];
}

// W3: f32 [J][128][6] -> f16 swizzled [J][kc(4)][lane][e], cols 6..15 zero
__global__ void cvt_w3(const float* __restrict__ src, _Float16* __restrict__ dst,
                       int total) {
  int tid = blockIdx.x * blockDim.x + threadIdx.x;
  if (tid >= total) return;
  int j      = tid / (HID * OUTP);
  int d      = tid - j * (HID * OUTP);
  int kc     = d >> 9;
  int within = d & 511;
  int lane   = within >> 4;
  int e      = within & 15;
  int lo = lane & 15, hi = lane >> 4;
  int kin = (e < 8) ? (hi * 8 + e) : (16 + hi * 8 + (e - 8));
  int k  = kc * 32 + kin;
  float v = (lo < OUTD) ? src[((size_t)j * HID + k) * OUTD + lo] : 0.0f;
  dst[tid] = (_Float16)v;
}

// ---------------------------------------------------------------------------
// gfx1250 async global->LDS DMA (ASYNCcnt-tracked, CDNA5 ISA).
// LDS operand: low 32 bits of the generic shared pointer == wave-relative
// LDS byte offset (flat LDS addresses truncate to the DS offset).
// ---------------------------------------------------------------------------
__device__ __forceinline__ void async_b128(const void* gsrc, const void* ldst) {
  asm volatile("global_load_async_to_lds_b128 %0, %1, off"
               :: "v"((unsigned)(unsigned long long)(uintptr_t)ldst),
                  "v"((unsigned long long)(uintptr_t)gsrc)
               : "memory");
}
__device__ __forceinline__ void wait_async0() {
  asm volatile("s_wait_asynccnt 0x0" ::: "memory");
}
// stage n16 * 16 bytes, cooperatively across 64 threads
__device__ __forceinline__ void stage_w_async(const _Float16* src, _Float16* dst,
                                              int n16, int t) {
  for (int i = t; i < n16; i += 64)
    async_b128((const char*)src + (size_t)i * 16, (char*)dst + (size_t)i * 16);
}

// ---------------------------------------------------------------------------
// A-fragment load: two contiguous 8-half runs (k..k+7 and k+16..k+23)
// ---------------------------------------------------------------------------
__device__ __forceinline__ v16h load_a_frag(const _Float16* p) {
  v8h a0 = *(const v8h*)p;
  v8h a1 = *(const v8h*)(p + 16);
  v16h a;
#pragma unroll
  for (int e = 0; e < 8; ++e) { a[e] = a0[e]; a[e + 8] = a1[e]; }
  return a;
}

__device__ __forceinline__ v8f zero_v8f() {
  v8f z = {0.f, 0.f, 0.f, 0.f, 0.f, 0.f, 0.f, 0.f};
  return z;
}

// ---------------------------------------------------------------------------
// Main fused kernel: one block = 32 batch rows x 1 joint, 64 threads (2 waves)
// ---------------------------------------------------------------------------
__global__ __launch_bounds__(64)
void posemlp_fused(const float* __restrict__ X,     // [B][24][256]
                   const _Float16* __restrict__ Wq, // swizzled f16 weights
                   const float* __restrict__ b1,    // [24][128]
                   const float* __restrict__ b2,    // [24][128]
                   const float* __restrict__ b3,    // [24][6]
                   float* __restrict__ out)         // [B][24][6]
{
  __shared__ _Float16 sX[MT * SA];   // input acts (f16)           16896 B
  __shared__ _Float16 sW[HID * HID]; // weight stage (swizzled)    32768 B
  __shared__ _Float16 sH[MT * SH];   // hidden acts (f16)           8704 B

  const int j    = blockIdx.y;
  const int row0 = blockIdx.x * MT;
  const int t    = threadIdx.x;        // 0..63
  const int lane = t & 31;
  const int wave = t >> 5;             // 0..1
  const int lo   = lane & 15;
  const int hi   = lane >> 4;
  const int mloc = wave * 16;          // this wave's 16-row slice

  const _Float16* W1s = Wq + (size_t)j * IND * HID;

  // ---- issue async DMA of W1 chunk 0 first: overlaps with input convert
  stage_w_async(W1s, sW, 2048, t);

  // ---- stage input tile: 32 rows x 256 f32 -> f16 (coalesced b128 loads)
#pragma unroll 4
  for (int i = t; i < MT * IND / 4; i += 64) {
    int r  = i >> 6;                   // 64 float4 per row
    int c4 = i & 63;
    float4 v = ((const float4*)X)[((size_t)(row0 + r) * NJ + j) * (IND / 4) + c4];
    int c = c4 * 4;
    sX[r * SA + c + 0] = (_Float16)v.x;
    sX[r * SA + c + 1] = (_Float16)v.y;
    sX[r * SA + c + 2] = (_Float16)v.z;
    sX[r * SA + c + 3] = (_Float16)v.w;
  }

  // ======================= Layer 1: 256 -> 128 ============================
  v8f acc[8];
#pragma unroll
  for (int nt = 0; nt < 8; ++nt) acc[nt] = zero_v8f();

  for (int c = 0; c < 2; ++c) {        // two 128-deep K chunks of W1
    if (c) {                           // restage: all waves done with chunk 0
      __syncthreads();
      stage_w_async(W1s + 16384, sW, 2048, t);
    }
    wait_async0();                     // my DMAs landed in LDS
    __syncthreads();                   // everyone's DMAs + sX stores visible
#pragma unroll
    for (int kcl = 0; kcl < 4; ++kcl) {
      const int kb = c * 128 + kcl * 32;
      v16h a = load_a_frag(&sX[(mloc + lo) * SA + kb + hi * 8]);
#pragma unroll
      for (int nt = 0; nt < 8; ++nt) {
        v16h b = *(const v16h*)&sW[(kcl * 8 + nt) * 512 + lane * 16];
        acc[nt] = __builtin_amdgcn_wmma_f32_16x16x32_f16(
            false, a, false, b, (short)0, acc[nt], false, false);
      }
    }
  }

  // all waves finished reading sW -> start W2 DMA, overlap with epilogue
  __syncthreads();
  stage_w_async(Wq + OFF_W2 + (size_t)j * HID * HID, sW, 2048, t);

  // epilogue: +b1, relu, f16 -> sH (overlaps W2 DMA)
#pragma unroll
  for (int nt = 0; nt < 8; ++nt) {
    int n = nt * 16 + lo;
    float bias = b1[j * HID + n];
#pragma unroll
    for (int r = 0; r < 8; ++r) {
      float v = acc[nt][r] + bias;
      v = v > 0.f ? v : 0.f;
      int m = mloc + hi * 8 + r;
      sH[m * SH + n] = (_Float16)v;
    }
  }

  // ======================= Layer 2: 128 -> 128 ============================
  wait_async0();
  __syncthreads();                     // W2 in LDS + all sH writes visible

  v8f acc2[8];
#pragma unroll
  for (int nt = 0; nt < 8; ++nt) acc2[nt] = zero_v8f();

#pragma unroll
  for (int kc = 0; kc < 4; ++kc) {
    const int kb = kc * 32;
    v16h a = load_a_frag(&sH[(mloc + lo) * SH + kb + hi * 8]);
#pragma unroll
    for (int nt = 0; nt < 8; ++nt) {
      v16h b = *(const v16h*)&sW[(kc * 8 + nt) * 512 + lane * 16];
      acc2[nt] = __builtin_amdgcn_wmma_f32_16x16x32_f16(
          false, a, false, b, (short)0, acc2[nt], false, false);
    }
  }

  // all waves finished reading sW -> start W3 DMA, overlap with epilogue
  __syncthreads();
  stage_w_async(Wq + OFF_W3 + (size_t)j * HID * OUTP, sW, 256, t);

  // epilogue: +b2, relu, f16 -> sH in place (wave touches only its own rows)
#pragma unroll
  for (int nt = 0; nt < 8; ++nt) {
    int n = nt * 16 + lo;
    float bias = b2[j * HID + n];
#pragma unroll
    for (int r = 0; r < 8; ++r) {
      float v = acc2[nt][r] + bias;
      v = v > 0.f ? v : 0.f;
      int m = mloc + hi * 8 + r;
      sH[m * SH + n] = (_Float16)v;
    }
  }

  // ======================= Layer 3: 128 -> 6 (padded 16) ==================
  wait_async0();
  __syncthreads();

  v8f acc3 = zero_v8f();
#pragma unroll
  for (int kc = 0; kc < 4; ++kc) {
    const int kb = kc * 32;
    v16h a = load_a_frag(&sH[(mloc + lo) * SH + kb + hi * 8]);
    v16h b = *(const v16h*)&sW[kc * 512 + lane * 16];
    acc3 = __builtin_amdgcn_wmma_f32_16x16x32_f16(
        false, a, false, b, (short)0, acc3, false, false);
  }

  // epilogue: +b3, store only n<6 (divergence strictly after last WMMA)
  if (lo < OUTD) {
    float bias = b3[j * OUTD + lo];
#pragma unroll
    for (int r = 0; r < 8; ++r) {
      int m = mloc + hi * 8 + r;
      out[((size_t)(row0 + m) * NJ + j) * OUTD + lo] = acc3[r] + bias;
    }
  }
}

// ---------------------------------------------------------------------------
extern "C" void kernel_launch(void* const* d_in, const int* in_sizes, int n_in,
                              void* d_out, int out_size, void* d_ws, size_t ws_size,
                              hipStream_t stream) {
  const float* pose = (const float*)d_in[0];
  const float* W1   = (const float*)d_in[1];
  const float* b1   = (const float*)d_in[2];
  const float* W2   = (const float*)d_in[3];
  const float* b2   = (const float*)d_in[4];
  const float* W3   = (const float*)d_in[5];
  const float* b3   = (const float*)d_in[6];
  float* out = (float*)d_out;

  _Float16* Wq = (_Float16*)d_ws;

  // weight pack (tiny: ~2.4 MB written, stays resident in L2 for main pass)
  {
    int total1 = (int)W1S_ELEMS;
    cvt_swz<<<(total1 + 255) / 256, 256, 0, stream>>>(W1, Wq, IND, HID,
                                                      IND * HID, total1);
    int total2 = (int)W2S_ELEMS;
    cvt_swz<<<(total2 + 255) / 256, 256, 0, stream>>>(W2, Wq + OFF_W2, HID, HID,
                                                      HID * HID, total2);
    int total3 = (int)W3S_ELEMS;
    cvt_w3<<<(total3 + 255) / 256, 256, 0, stream>>>(W3, Wq + OFF_W3, total3);
  }

  // fused MLP: one block per (32 batch rows, joint)
  dim3 grid(BATCH / MT, NJ);
  posemlp_fused<<<grid, 64, 0, stream>>>(pose, Wq, b1, b2, b3, out);
}
